// MyRGCN_88149908783511
// MI455X (gfx1250) — compile-verified, matrix operand order checked by reading, not run on hardware
//
#include <hip/hip_runtime.h>
#include <hip/hip_bf16.h>

#define N_NODES 50000
#define N_EDGES 1600000
#define F_IN 128
#define HIDDEN 128
#define N_REL 8
#define N_GRAPHS 64
#define K_FUSED (N_REL * F_IN + F_IN)   // 1152 = [8 relation means | root input]
#define KT_RGCN (K_FUSED / 32)          // 36 K-tiles of 32
#define KT_LIN (HIDDEN / 32)            // 4
#define NT_TILES (HIDDEN / 16)          // 8 N-tiles of 16
#define KCH (K_FUSED / 8)               // 144 8-wide chunks per A row

typedef __attribute__((ext_vector_type(16))) __bf16 v16bf;
typedef __attribute__((ext_vector_type(8))) float v8f;

struct U128 { unsigned int a, b, c, d; };
union Frag16 { v16bf v; unsigned short u[16]; U128 q[2]; };

__device__ __forceinline__ unsigned int f2bf_bits(float f) {
  union { float f; unsigned int u; } x;
  x.f = f;
  unsigned int u = x.u;
  u += 0x7FFFu + ((u >> 16) & 1u);   // round-to-nearest-even
  return u >> 16;
}
__device__ __forceinline__ unsigned short f2bf(float f) {
  return (unsigned short)f2bf_bits(f);
}
// pack two floats -> two bf16 in one dword (lo = a, hi = b)
__device__ __forceinline__ unsigned int pk_bf16(float a, float b) {
  return (f2bf_bits(a) & 0xFFFFu) | (f2bf_bits(b) << 16);
}

// ---------------------------------------------------------------- utility ---
__global__ void k_zero4(float4* p, long n4) {
  long i = (long)blockIdx.x * blockDim.x + threadIdx.x;
  if (i < n4) p[i] = make_float4(0.f, 0.f, 0.f, 0.f);
}

// edge_type[e] = argmax(edge_attr[e, 0..7])
__global__ void k_edge_type(const float* __restrict__ edge_attr,
                            int* __restrict__ etype, int n) {
  int e = blockIdx.x * blockDim.x + threadIdx.x;
  if (e >= n) return;
  const float* a = edge_attr + (size_t)e * N_REL;
  int bi = 0; float bv = a[0];
#pragma unroll
  for (int i = 1; i < N_REL; ++i) { float v = a[i]; if (v > bv) { bv = v; bi = i; } }
  etype[e] = bi;
}

__global__ void k_count(const int* __restrict__ dstv, const int* __restrict__ etype,
                        float* __restrict__ cnt, int n) {
  int e = blockIdx.x * blockDim.x + threadIdx.x;
  if (e >= n) return;
  atomicAdd(cnt + (size_t)dstv[e] * N_REL + etype[e], 1.0f);
}

__global__ void k_recip(float* __restrict__ c, int n) {
  int i = blockIdx.x * blockDim.x + threadIdx.x;
  if (i < n) c[i] = 1.0f / fmaxf(c[i], 1.0f);
}

// ---------------------------------------------------- weight pre-packing ---
// Logical B matrix is K x 128 (rows 0..split-1 from pa, rest from pb), packed
// into v_wmma_f32_16x16x32_bf16 B-fragment layout: per (kt, nt, lane) 16 bf16
// values, contiguous 32 B per lane. Lane l: n = nt*16 + (l&15), half = l>>4.
// VGPR v in [0,3]: K = kt*32 + half*8 + 2v + {0,1};  v in [4,7]: +16.
__global__ void k_pack_b(const float* __restrict__ pa, const float* __restrict__ pb,
                         int split, int kt_total, unsigned short* __restrict__ out) {
  int t = blockIdx.x * blockDim.x + threadIdx.x;
  if (t >= kt_total * NT_TILES * 32) return;
  int lane = t & 31;
  int nt = (t >> 5) & (NT_TILES - 1);
  int kt = t >> 8;                       // NT_TILES*32 == 256
  int half = lane >> 4;
  int n = nt * 16 + (lane & 15);
  unsigned short* d = out + ((size_t)(kt * NT_TILES + nt) * 32 + lane) * 16;
#pragma unroll
  for (int idx = 0; idx < 16; ++idx) {
    int kk = (idx < 8) ? (half * 8 + idx) : (16 + half * 8 + (idx - 8));
    int k = kt * 32 + kk;
    float f = (k < split) ? pa[(size_t)k * HIDDEN + n]
                          : pb[(size_t)(k - split) * HIDDEN + n];
    d[idx] = f2bf(f);
  }
}

// ------------------------------------------------------------ aggregation ---
// One wave per edge; lane handles 4 features (float4 gather, 4 f32 atomics).
__global__ void k_scatter(const int* __restrict__ srcv, const int* __restrict__ dstv,
                          const int* __restrict__ etype, const float* __restrict__ h,
                          float* __restrict__ agg, int n) {
  int w = (int)(((long)blockIdx.x * blockDim.x + threadIdx.x) >> 5);
  int lane = threadIdx.x & 31;
  if (w >= n) return;
  int s = srcv[w], d = dstv[w], t = etype[w];
  float4 v = *(const float4*)(h + (size_t)s * F_IN + lane * 4);
  float* b = agg + (((size_t)d * N_REL + t) << 7) + lane * 4;
  atomicAdd(b + 0, v.x); atomicAdd(b + 1, v.y);
  atomicAdd(b + 2, v.z); atomicAdd(b + 3, v.w);
}

__global__ void k_pool_count(const int* __restrict__ batch, float* __restrict__ pc, int n) {
  int i = blockIdx.x * blockDim.x + threadIdx.x;
  if (i < n) atomicAdd(pc + batch[i], 1.0f);
}

__global__ void k_pool_scatter(const int* __restrict__ batch, const float* __restrict__ h,
                               float* __restrict__ ps, int n) {
  int w = (int)(((long)blockIdx.x * blockDim.x + threadIdx.x) >> 5);
  int lane = threadIdx.x & 31;
  if (w >= n) return;
  int g = batch[w];
  float4 v = *(const float4*)(h + (size_t)w * HIDDEN + lane * 4);
  float* b = ps + (size_t)g * HIDDEN + lane * 4;
  atomicAdd(b + 0, v.x); atomicAdd(b + 1, v.y);
  atomicAdd(b + 2, v.z); atomicAdd(b + 3, v.w);
}

// ---------------------------------------------------------------- WMMA GEMM -
// Per block: stage the 16 x 1152 A row-block (normalized, bf16) in LDS once,
// then 8 waves each compute one 16x16 C tile. The agg-vs-h select and the
// mean scaling live only in the staging pass, so the MAC loop is just
// 2x ds_load_b128 (A frag) + 2x global_load_b128 (packed B) + v_wmma.
__global__ void __launch_bounds__(256) k_rgcn_gemm(
    const float* __restrict__ agg, const float* __restrict__ cnt_inv,
    const float* __restrict__ h_in, const unsigned short* __restrict__ bpack,
    const float* __restrict__ bias, float* __restrict__ h_out) {
  __shared__ unsigned short As[16 * K_FUSED];   // 36,864 B
  int mt = blockIdx.x;
  int tid = threadIdx.x;

  // ---- stage A block: 16 rows x 144 chunks of 8 f32 -> bf16 ----
  for (int c = tid; c < 16 * KCH; c += 256) {
    int r = c / KCH;                 // 0..15
    int col = (c - r * KCH) * 8;     // 0..1144, 8-aligned
    int row = mt * 16 + r;
    const float* sp; float s;
    if (col < N_REL * F_IN) {
      int rel = col >> 7;
      sp = agg + (((size_t)row * N_REL + rel) << 7) + (col & 127);
      s = cnt_inv[(size_t)row * N_REL + rel];
    } else {
      sp = h_in + ((size_t)row << 7) + (col - N_REL * F_IN);
      s = 1.0f;
    }
    float4 f0 = *(const float4*)sp;
    float4 f1 = *(const float4*)(sp + 4);
    U128 o;
    o.a = pk_bf16(f0.x * s, f0.y * s);
    o.b = pk_bf16(f0.z * s, f0.w * s);
    o.c = pk_bf16(f1.x * s, f1.y * s);
    o.d = pk_bf16(f1.z * s, f1.w * s);
    *(U128*)(As + r * K_FUSED + col) = o;
  }
  __syncthreads();

  // ---- MAC loop ----
  int nt = tid >> 5;
  int lane = tid & 31;
  int half = lane >> 4;
  int l15 = lane & 15;
  v8f acc = {0.f, 0.f, 0.f, 0.f, 0.f, 0.f, 0.f, 0.f};
  const U128* bbase = (const U128*)bpack;
  const unsigned short* arow = As + l15 * K_FUSED + half * 8;
  for (int kt = 0; kt < KT_RGCN; ++kt) {
    Frag16 a, b;
    const unsigned short* ap = arow + kt * 32;
    a.q[0] = *(const U128*)ap;          // K offsets half*8 + 0..7
    a.q[1] = *(const U128*)(ap + 16);   // K offsets 16 + half*8 + 0..7
    const U128* bp = bbase + ((size_t)(kt * NT_TILES + nt) * 32 + lane) * 2;
    b.q[0] = bp[0]; b.q[1] = bp[1];
    acc = __builtin_amdgcn_wmma_f32_16x16x32_bf16(false, a.v, false, b.v,
                                                  (short)0, acc, false, false);
  }
  int col = nt * 16 + l15;
  float bv = bias[col];
#pragma unroll
  for (int j = 0; j < 8; ++j) {
    float o = fmaxf(acc[j] + bv, 0.0f);                 // ReLU
    h_out[(size_t)(mt * 16 + half * 8 + j) * HIDDEN + col] = o;
  }
}

// pooled[64,128] @ lin_w + lin_b -> d_out (no ReLU); mean applied on A load.
__global__ void __launch_bounds__(256) k_pool_gemm(
    const float* __restrict__ ps, const float* __restrict__ pc,
    const unsigned short* __restrict__ bpack, const float* __restrict__ bias,
    float* __restrict__ out) {
  int mt = blockIdx.x;
  int nt = threadIdx.x >> 5;
  int lane = threadIdx.x & 31;
  int half = lane >> 4;
  int l15 = lane & 15;
  int g = mt * 16 + l15;
  float s = 1.0f / fmaxf(pc[g], 1.0f);
  v8f acc = {0.f, 0.f, 0.f, 0.f, 0.f, 0.f, 0.f, 0.f};
  const U128* bbase = (const U128*)bpack;
#pragma unroll
  for (int kt = 0; kt < KT_LIN; ++kt) {
    Frag16 a, b;
#pragma unroll
    for (int ch = 0; ch < 2; ++ch) {
      int c = kt * 32 + half * 8 + ch * 16;
      const float* sp = ps + ((size_t)g << 7) + c;
      float4 f0 = *(const float4*)sp;
      float4 f1 = *(const float4*)(sp + 4);
      unsigned short* u = a.u + ch * 8;
      u[0] = f2bf(f0.x * s); u[1] = f2bf(f0.y * s);
      u[2] = f2bf(f0.z * s); u[3] = f2bf(f0.w * s);
      u[4] = f2bf(f1.x * s); u[5] = f2bf(f1.y * s);
      u[6] = f2bf(f1.z * s); u[7] = f2bf(f1.w * s);
    }
    const U128* bp = bbase + ((size_t)(kt * NT_TILES + nt) * 32 + lane) * 2;
    b.q[0] = bp[0]; b.q[1] = bp[1];
    acc = __builtin_amdgcn_wmma_f32_16x16x32_bf16(false, a.v, false, b.v,
                                                  (short)0, acc, false, false);
  }
  int col = nt * 16 + l15;
  float bv = bias[col];
#pragma unroll
  for (int j = 0; j < 8; ++j)
    out[(size_t)(mt * 16 + half * 8 + j) * HIDDEN + col] = acc[j] + bv;
}

// -------------------------------------------------------------------- host ---
extern "C" void kernel_launch(void* const* d_in, const int* in_sizes, int n_in,
                              void* d_out, int out_size, void* d_ws, size_t ws_size,
                              hipStream_t stream) {
  const float* x         = (const float*)d_in[0];
  const int*   ei        = (const int*)d_in[1];
  const float* edge_attr = (const float*)d_in[2];
  const int*   batch     = (const int*)d_in[3];
  const float* W[3]    = {(const float*)d_in[4], (const float*)d_in[7], (const float*)d_in[10]};
  const float* root[3] = {(const float*)d_in[5], (const float*)d_in[8], (const float*)d_in[11]};
  const float* bias[3] = {(const float*)d_in[6], (const float*)d_in[9], (const float*)d_in[12]};
  const float* lin_w = (const float*)d_in[13];
  const float* lin_b = (const float*)d_in[14];
  float* out = (float*)d_out;
  (void)in_sizes; (void)n_in; (void)out_size; (void)ws_size;

  char* ws = (char*)d_ws;
  size_t off = 0;
  #define WS_ALLOC(ty, nbytes) (ty*)(ws + off); off = (off + (size_t)(nbytes) + 255) & ~(size_t)255
  int*   etype = WS_ALLOC(int,   (size_t)N_EDGES * 4);
  float* cnt   = WS_ALLOC(float, (size_t)N_NODES * N_REL * 4);          // -> reciprocals
  float* agg   = WS_ALLOC(float, (size_t)N_NODES * N_REL * F_IN * 4);   // 204.8 MB
  float* hA    = WS_ALLOC(float, (size_t)N_NODES * HIDDEN * 4);
  float* hB    = WS_ALLOC(float, (size_t)N_NODES * HIDDEN * 4);
  unsigned short* bp0 = WS_ALLOC(unsigned short, (size_t)KT_RGCN * NT_TILES * 32 * 16 * 2);
  unsigned short* bp1 = WS_ALLOC(unsigned short, (size_t)KT_RGCN * NT_TILES * 32 * 16 * 2);
  unsigned short* bp2 = WS_ALLOC(unsigned short, (size_t)KT_RGCN * NT_TILES * 32 * 16 * 2);
  unsigned short* bpl = WS_ALLOC(unsigned short, (size_t)KT_LIN * NT_TILES * 32 * 16 * 2);
  float* psum = WS_ALLOC(float, (size_t)N_GRAPHS * HIDDEN * 4);
  float* pcnt = WS_ALLOC(float, (size_t)N_GRAPHS * 4);
  #undef WS_ALLOC
  unsigned short* bp[3] = {bp0, bp1, bp2};

  const int* srcv = ei;
  const int* dstv = ei + N_EDGES;

  // one-time: edge types, relation counts -> reciprocals, weight packing
  k_edge_type<<<(N_EDGES + 255) / 256, 256, 0, stream>>>(edge_attr, etype, N_EDGES);
  long cnt4 = (long)N_NODES * N_REL / 4;
  k_zero4<<<(int)((cnt4 + 255) / 256), 256, 0, stream>>>((float4*)cnt, cnt4);
  k_count<<<(N_EDGES + 255) / 256, 256, 0, stream>>>(dstv, etype, cnt, N_EDGES);
  k_recip<<<(N_NODES * N_REL + 255) / 256, 256, 0, stream>>>(cnt, N_NODES * N_REL);
  int pth = KT_RGCN * NT_TILES * 32;
  for (int l = 0; l < 3; ++l)
    k_pack_b<<<(pth + 255) / 256, 256, 0, stream>>>(W[l], root[l], N_REL * F_IN, KT_RGCN, bp[l]);
  int lth = KT_LIN * NT_TILES * 32;
  k_pack_b<<<(lth + 255) / 256, 256, 0, stream>>>(nullptr, lin_w, 0, KT_LIN, bpl);

  // 3 RGCN layers
  const float* hin = x;
  float* hout = hA;
  long agg4 = (long)N_NODES * N_REL * F_IN / 4;
  for (int l = 0; l < 3; ++l) {
    k_zero4<<<(int)((agg4 + 255) / 256), 256, 0, stream>>>((float4*)agg, agg4);
    k_scatter<<<(int)(((long)N_EDGES * 32 + 255) / 256), 256, 0, stream>>>(
        srcv, dstv, etype, hin, agg, N_EDGES);
    k_rgcn_gemm<<<N_NODES / 16, 256, 0, stream>>>(agg, cnt, hin, bp[l], bias[l], hout);
    hin = hout;
    hout = (hout == hA) ? hB : hA;
  }

  // global mean pool + linear head
  long p4 = (long)N_GRAPHS * HIDDEN / 4;
  k_zero4<<<(int)((p4 + 255) / 256), 256, 0, stream>>>((float4*)psum, p4);
  k_zero4<<<1, 256, 0, stream>>>((float4*)pcnt, N_GRAPHS / 4);
  k_pool_count<<<(N_NODES + 255) / 256, 256, 0, stream>>>(batch, pcnt, N_NODES);
  k_pool_scatter<<<(int)(((long)N_NODES * 32 + 255) / 256), 256, 0, stream>>>(
      batch, hin, psum, N_NODES);
  k_pool_gemm<<<N_GRAPHS / 16, 256, 0, stream>>>(psum, pcnt, bpl, lin_b, out);
}
